// SimpleTransformerBlock_88905823027203
// MI455X (gfx1250) — matrix operand order repaired and checked by reading, not compiled, not run on hardware
//
#include <hip/hip_runtime.h>
#include <hip/hip_bf16.h>

#define D_MODEL 1024
#define HEADS   16
#define HEAD_DIM 64
#define SEQ     2048
#define BATCH   4
#define ROWS    (BATCH * SEQ)   // 8192

typedef __attribute__((ext_vector_type(16))) __bf16 v16bf;
typedef __attribute__((ext_vector_type(8)))  float  v8f;

union FragBF {
    v16bf  v;
    uint4  q[2];
    __bf16 e[16];
};

__device__ inline v8f wmma_bf16(const v16bf& a, const v16bf& b, const v8f& c) {
    // D = A(16x32 bf16) * B(32x16 bf16) + C(16x16 f32)
    return __builtin_amdgcn_wmma_f32_16x16x32_bf16(
        /*neg_a=*/false, a, /*neg_b=*/false, b,
        /*c_mod=*/(short)0, c, /*reuse_a=*/false, /*reuse_b=*/false);
}

// async global->LDS copy of 16 bytes (tracked by ASYNCcnt)
__device__ inline void async_copy_b128(const void* gaddr, unsigned lds_addr) {
    asm volatile("global_load_async_to_lds_b128 %0, %1, off"
                 :: "v"(lds_addr), "v"((unsigned long long)(size_t)gaddr)
                 : "memory");
}
__device__ inline void wait_async0() {
    asm volatile("s_wait_asynccnt 0" ::: "memory");
}
__device__ inline unsigned lds_off(const void* p) {
    // generic LDS pointer: low 32 bits are the LDS byte address (flat aperture rule)
    return (unsigned)(size_t)p;
}

// ---------------------------------------------------------------------------
// fp32 -> bf16 elementwise convert (weights)
// ---------------------------------------------------------------------------
__global__ void f32_to_bf16_kernel(const float* __restrict__ in,
                                   __bf16* __restrict__ out, int n) {
    int i = blockIdx.x * blockDim.x + threadIdx.x;
    if (i < n) out[i] = (__bf16)in[i];
}

// ---------------------------------------------------------------------------
// LayerNorm: one block per row of D_MODEL, fp32 in, bf16 out
// ---------------------------------------------------------------------------
__global__ __launch_bounds__(256) void layernorm_bf16_kernel(
    const float* __restrict__ x, const float* __restrict__ g,
    const float* __restrict__ b, __bf16* __restrict__ out) {
    const int row = blockIdx.x;
    const float* xr = x + (size_t)row * D_MODEL;

    float vals[4], s = 0.f, ss = 0.f;
#pragma unroll
    for (int j = 0; j < 4; j++) {
        float v = xr[threadIdx.x + 256 * j];
        vals[j] = v; s += v; ss += v * v;
    }
    __shared__ float r1[256], r2[256];
    r1[threadIdx.x] = s; r2[threadIdx.x] = ss;
    __syncthreads();
    for (int off = 128; off > 0; off >>= 1) {
        if (threadIdx.x < off) {
            r1[threadIdx.x] += r1[threadIdx.x + off];
            r2[threadIdx.x] += r2[threadIdx.x + off];
        }
        __syncthreads();
    }
    const float mu  = r1[0] * (1.0f / D_MODEL);
    const float var = r2[0] * (1.0f / D_MODEL) - mu * mu;
    const float rs  = rsqrtf(var + 1e-5f);
#pragma unroll
    for (int j = 0; j < 4; j++) {
        int c = threadIdx.x + 256 * j;
        out[(size_t)row * D_MODEL + c] = (__bf16)((vals[j] - mu) * rs * g[c] + b[c]);
    }
}

// ---------------------------------------------------------------------------
// Double-buffered WMMA GEMM: C = A(MxK bf16) @ B(KxN bf16) + bias.
// Block 256 = 8 waves; tile 128x64; BK = 64 (two WMMA K-steps per stage).
// A tile streams global->LDS with GLOBAL_LOAD_ASYNC_TO_LDS_B128 (ASYNCcnt);
// B tile is software-pipelined through VGPRs and scatter-transposed into LDS.
//   EPI 0: bias, store bf16
//   EPI 1: bias + resid, store f32
//   EPI 2: bias + exact GELU, store bf16
// ---------------------------------------------------------------------------
template <int EPI>
__global__ __launch_bounds__(256) void gemm_wmma_kernel(
    const __bf16* __restrict__ A, const __bf16* __restrict__ B,
    const float* __restrict__ bias, const float* __restrict__ resid,
    void* __restrict__ C, int M, int N, int K) {
    __shared__ __bf16 As[2][128][72];   // row-major, 144B rows (16B aligned)
    __shared__ __bf16 Bs[2][64][72];    // transposed: Bs[buf][n][k]

    const int tid  = threadIdx.x;
    const int wave = tid >> 5, lane = tid & 31;
    const int half = lane >> 4, l16 = lane & 15;
    const int tileM = blockIdx.y * 128, tileN = blockIdx.x * 64;

    v8f acc[4] = {{}, {}, {}, {}};

    const int arow = tid >> 1;          // 0..127
    const int acol = (tid & 1) * 32;    // 0 / 32
    const int bk   = tid >> 2;          // 0..63
    const int bn   = (tid & 3) * 16;    // 0,16,32,48

    // ---- staging helpers --------------------------------------------------
    auto loadA_async = [&](int buf, int k0) {
        const __bf16* ap = A + (size_t)(tileM + arow) * K + k0 + acol;
#pragma unroll
        for (int c = 0; c < 4; c++)
            async_copy_b128(ap + c * 8, lds_off(&As[buf][arow][acol + c * 8]));
    };
    auto loadB_regs = [&](int k0, uint4& r0, uint4& r1) {
        const __bf16* bp = B + (size_t)(k0 + bk) * N + tileN + bn;
        r0 = *(const uint4*)bp;
        r1 = *(const uint4*)(bp + 8);
    };
    auto scatterB = [&](int buf, const uint4& r0, const uint4& r1) {
        FragBF f; f.q[0] = r0; f.q[1] = r1;
#pragma unroll
        for (int j = 0; j < 16; j++) Bs[buf][bn + j][bk] = f.e[j];
    };

    // ---- prologue: stage tile 0 ------------------------------------------
    uint4 br0, br1;
    loadA_async(0, 0);
    loadB_regs(0, br0, br1);
    scatterB(0, br0, br1);
    wait_async0();
    __syncthreads();

    int buf = 0;
    for (int k0 = 0; k0 < K; k0 += 64) {
        const bool prefetch = (k0 + 64) < K;
        if (prefetch) {
            loadA_async(buf ^ 1, k0 + 64);   // async copy overlaps the WMMAs
            loadB_regs(k0 + 64, br0, br1);   // global latency hidden under compute
        }

        const int mrow = wave * 16 + l16;
#pragma unroll
        for (int s = 0; s < 2; s++) {        // two K=32 WMMA steps per stage
            FragBF a;
            a.q[0] = *(const uint4*)&As[buf][mrow][s * 32 + half * 8];
            a.q[1] = *(const uint4*)&As[buf][mrow][s * 32 + 16 + half * 8];
#pragma unroll
            for (int t = 0; t < 4; t++) {
                FragBF bf;
                const int nc = t * 16 + l16;
                bf.q[0] = *(const uint4*)&Bs[buf][nc][s * 32 + half * 8];
                bf.q[1] = *(const uint4*)&Bs[buf][nc][s * 32 + 16 + half * 8];
                acc[t] = wmma_bf16(a.v, bf.v, acc[t]);
            }
        }

        if (prefetch) scatterB(buf ^ 1, br0, br1);
        wait_async0();
        __syncthreads();
        buf ^= 1;
    }

    // ---- epilogue ---------------------------------------------------------
#pragma unroll
    for (int t = 0; t < 4; t++) {
#pragma unroll
        for (int i = 0; i < 8; i++) {
            const int row = tileM + wave * 16 + i + half * 8;  // C/D layout
            const int col = tileN + t * 16 + l16;
            float v = acc[t][i] + bias[col];
            const size_t idx = (size_t)row * N + col;
            if (EPI == 0) {
                ((__bf16*)C)[idx] = (__bf16)v;
            } else if (EPI == 1) {
                ((float*)C)[idx] = v + resid[idx];
            } else {
                v = 0.5f * v * (1.0f + erff(v * 0.70710678118654752f));
                ((__bf16*)C)[idx] = (__bf16)v;
            }
        }
    }
}

// ---------------------------------------------------------------------------
// Flash attention: block = 128 threads = 4 waves, 64 query rows per (b,h).
// qkv is bf16 [ROWS, 3*D_MODEL]; out is bf16 [ROWS, D_MODEL].
// ---------------------------------------------------------------------------
__global__ __launch_bounds__(128) void attn_wmma_kernel(
    const __bf16* __restrict__ qkv, __bf16* __restrict__ out) {
    __shared__ __bf16 Vl[64][40];        // [hd][kv], transposed stage
    __shared__ __bf16 Pl[4][16][40];     // per-wave P tile [qrow][kv]

    const int tid  = threadIdx.x;
    const int wave = tid >> 5, lane = tid & 31;
    const int half = lane >> 4, l16 = lane & 15;
    const int h = blockIdx.y, b = blockIdx.z;
    const size_t rowbase = (size_t)b * SEQ;
    const int q0   = blockIdx.x * 64 + wave * 16;
    const int qoff = h * HEAD_DIM;
    const int koff = D_MODEL + h * HEAD_DIM;
    const int voff = 2 * D_MODEL + h * HEAD_DIM;
    const int LDQ  = 3 * D_MODEL;

    // Q fragments: hd 0..31 (lo) and 32..63 (hi); lane holds k = 8h..8h+7 and 16+8h..16+8h+7.
    FragBF qlo, qhi;
    {
        const __bf16* qp = qkv + (rowbase + q0 + l16) * LDQ + qoff;
        qlo.q[0] = *(const uint4*)(qp + half * 8);
        qlo.q[1] = *(const uint4*)(qp + 16 + half * 8);
        qhi.q[0] = *(const uint4*)(qp + 32 + half * 8);
        qhi.q[1] = *(const uint4*)(qp + 32 + 16 + half * 8);
    }

    v8f o0 = {}, o1 = {}, o2 = {}, o3 = {};
    float m[8], l[8];
#pragma unroll
    for (int i = 0; i < 8; i++) { m[i] = -1e30f; l[i] = 0.f; }

    const int vkv = tid >> 2;           // 0..31
    const int vhd = (tid & 3) * 16;     // 0,16,32,48

    for (int kv0 = 0; kv0 < SEQ; kv0 += 32) {
        __syncthreads();
        {   // cooperative V stage: Vl[hd][kv]
            const __bf16* vp = qkv + (rowbase + kv0 + vkv) * LDQ + voff + vhd;
            FragBF f;
            f.q[0] = *(const uint4*)vp;
            f.q[1] = *(const uint4*)(vp + 8);
#pragma unroll
            for (int j = 0; j < 16; j++) Vl[vhd + j][vkv] = f.e[j];
        }
        __syncthreads();

        // scores: two 16-col tiles, K-dim = hd = 64 -> 2 chained WMMAs each
        v8f s[2];
#pragma unroll
        for (int t = 0; t < 2; t++) {
            const __bf16* kp = qkv + (rowbase + kv0 + t * 16 + l16) * LDQ + koff;
            FragBF klo, khi;
            klo.q[0] = *(const uint4*)(kp + half * 8);
            klo.q[1] = *(const uint4*)(kp + 16 + half * 8);
            khi.q[0] = *(const uint4*)(kp + 32 + half * 8);
            khi.q[1] = *(const uint4*)(kp + 32 + 16 + half * 8);
            v8f c = {};
            c = wmma_bf16(qlo.v, klo.v, c);
            c = wmma_bf16(qhi.v, khi.v, c);
            s[t] = c;
        }

        // online softmax per accumulator slot (row = i + half*8)
#pragma unroll
        for (int i = 0; i < 8; i++) {
            float s0 = s[0][i] * 0.125f;   // 1/sqrt(64)
            float s1 = s[1][i] * 0.125f;
            float tm = fmaxf(s0, s1);
#pragma unroll
            for (int x = 1; x < 16; x <<= 1) tm = fmaxf(tm, __shfl_xor(tm, x, 16));
            const float mnew = fmaxf(m[i], tm);
            const float corr = __expf(m[i] - mnew);
            const float p0 = __expf(s0 - mnew);
            const float p1 = __expf(s1 - mnew);
            float rs = p0 + p1;
#pragma unroll
            for (int x = 1; x < 16; x <<= 1) rs += __shfl_xor(rs, x, 16);
            l[i] = l[i] * corr + rs;
            m[i] = mnew;
            o0[i] *= corr; o1[i] *= corr; o2[i] *= corr; o3[i] *= corr;
            const int prow = i + half * 8;
            Pl[wave][prow][l16]      = (__bf16)p0;
            Pl[wave][prow][16 + l16] = (__bf16)p1;
        }
        asm volatile("s_wait_dscnt 0" ::: "memory");

        // P as A-fragment (D-layout -> A-layout via LDS)
        FragBF p;
        p.q[0] = *(const uint4*)&Pl[wave][l16][half * 8];
        p.q[1] = *(const uint4*)&Pl[wave][l16][16 + half * 8];

#pragma unroll
        for (int t = 0; t < 4; t++) {
            FragBF vf;
            const int hd = t * 16 + l16;
            vf.q[0] = *(const uint4*)&Vl[hd][half * 8];
            vf.q[1] = *(const uint4*)&Vl[hd][16 + half * 8];
            v8f& o = (t == 0) ? o0 : (t == 1) ? o1 : (t == 2) ? o2 : o3;
            o = wmma_bf16(p.v, vf.v, o);
        }
    }

    // normalize and store bf16 to [ROWS, D_MODEL]
#pragma unroll
    for (int t = 0; t < 4; t++) {
        const v8f& o = (t == 0) ? o0 : (t == 1) ? o1 : (t == 2) ? o2 : o3;
#pragma unroll
        for (int i = 0; i < 8; i++) {
            const int row = (int)rowbase + q0 + i + half * 8;
            const int col = h * HEAD_DIM + t * 16 + l16;
            out[(size_t)row * D_MODEL + col] = (__bf16)(o[i] / l[i]);
        }
    }
}

// ---------------------------------------------------------------------------
// Host launcher
// ---------------------------------------------------------------------------
extern "C" void kernel_launch(void* const* d_in, const int* in_sizes, int n_in,
                              void* d_out, int out_size, void* d_ws, size_t ws_size,
                              hipStream_t stream) {
    const float* x      = (const float*)d_in[0];
    const float* ln1_g  = (const float*)d_in[1];
    const float* ln1_b  = (const float*)d_in[2];
    const float* w_qkv  = (const float*)d_in[3];
    const float* b_qkv  = (const float*)d_in[4];
    const float* w_out  = (const float*)d_in[5];
    const float* b_out  = (const float*)d_in[6];
    const float* ln2_g  = (const float*)d_in[7];
    const float* ln2_b  = (const float*)d_in[8];
    const float* w_ffn1 = (const float*)d_in[9];
    const float* b_ffn1 = (const float*)d_in[10];
    const float* w_ffn2 = (const float*)d_in[11];
    const float* b_ffn2 = (const float*)d_in[12];
    float* outp = (float*)d_out;

    char* ws = (char*)d_ws;
    size_t off = 0;
    auto carve = [&](size_t bytes) -> void* {
        void* p = ws + off;
        off += (bytes + 255) & ~(size_t)255;
        return p;
    };
    __bf16* h1   = (__bf16*)carve((size_t)ROWS * D_MODEL * 2);
    __bf16* wq   = (__bf16*)carve((size_t)D_MODEL * 3 * D_MODEL * 2);
    __bf16* wo   = (__bf16*)carve((size_t)D_MODEL * D_MODEL * 2);
    __bf16* wf1  = (__bf16*)carve((size_t)D_MODEL * 4 * D_MODEL * 2);
    __bf16* wf2  = (__bf16*)carve((size_t)4 * D_MODEL * D_MODEL * 2);
    __bf16* qkvb = (__bf16*)carve((size_t)ROWS * 3 * D_MODEL * 2);
    __bf16* attb = (__bf16*)carve((size_t)ROWS * D_MODEL * 2);
    float*  x1   = (float*) carve((size_t)ROWS * D_MODEL * 4);
    __bf16* h2   = (__bf16*)carve((size_t)ROWS * D_MODEL * 2);
    __bf16* gb   = (__bf16*)carve((size_t)ROWS * 4 * D_MODEL * 2);

    int n;
    n = D_MODEL * 3 * D_MODEL;
    f32_to_bf16_kernel<<<(n + 255) / 256, 256, 0, stream>>>(w_qkv, wq, n);
    n = D_MODEL * D_MODEL;
    f32_to_bf16_kernel<<<(n + 255) / 256, 256, 0, stream>>>(w_out, wo, n);
    n = D_MODEL * 4 * D_MODEL;
    f32_to_bf16_kernel<<<(n + 255) / 256, 256, 0, stream>>>(w_ffn1, wf1, n);
    n = 4 * D_MODEL * D_MODEL;
    f32_to_bf16_kernel<<<(n + 255) / 256, 256, 0, stream>>>(w_ffn2, wf2, n);

    // LN1 -> h1
    layernorm_bf16_kernel<<<ROWS, 256, 0, stream>>>(x, ln1_g, ln1_b, h1);

    // qkv = h1 @ w_qkv + b_qkv   (bf16 out)
    gemm_wmma_kernel<0><<<dim3(3 * D_MODEL / 64, ROWS / 128), 256, 0, stream>>>(
        h1, wq, b_qkv, nullptr, qkvb, ROWS, 3 * D_MODEL, D_MODEL);

    // attention
    attn_wmma_kernel<<<dim3(SEQ / 64, HEADS, BATCH), 128, 0, stream>>>(qkvb, attb);

    // x1 = x + attn @ w_out + b_out   (f32 out)
    gemm_wmma_kernel<1><<<dim3(D_MODEL / 64, ROWS / 128), 256, 0, stream>>>(
        attb, wo, b_out, x, x1, ROWS, D_MODEL, D_MODEL);

    // LN2 -> h2
    layernorm_bf16_kernel<<<ROWS, 256, 0, stream>>>(x1, ln2_g, ln2_b, h2);

    // gb = gelu(h2 @ w_ffn1 + b_ffn1)   (bf16 out)
    gemm_wmma_kernel<2><<<dim3(4 * D_MODEL / 64, ROWS / 128), 256, 0, stream>>>(
        h2, wf1, b_ffn1, nullptr, gb, ROWS, 4 * D_MODEL, D_MODEL);

    // out = x1 + gb @ w_ffn2 + b_ffn2   (f32 out)
    gemm_wmma_kernel<1><<<dim3(D_MODEL / 64, ROWS / 128), 256, 0, stream>>>(
        gb, wf2, b_ffn2, x1, outp, ROWS, D_MODEL, 4 * D_MODEL);
}
